// PagedHSTUInferLayer_70617852281042
// MI455X (gfx1250) — compile-verified, matrix operand order checked by reading, not run on hardware
//
#include <hip/hip_runtime.h>

// ---------------------------------------------------------------------------
// HSTU inference layer for gfx1250 (MI455X), bf16 WMMA pipeline.
// B=8, S=2048, H=8, D=64, E=512, num_targets=64  (fixed by setup_inputs)
// ---------------------------------------------------------------------------

#define B_    8
#define S_    2048
#define H_    8
#define D_    64
#define E_    512
#define T_    (B_ * S_)        // 16384 rows
#define N4_   (4 * H_ * D_)    // 2048 fused columns
#define HIST_ (S_ - 64)        // 1984: keys below this are "history"

typedef __attribute__((ext_vector_type(16))) __bf16 v16bf;
typedef __attribute__((ext_vector_type(8)))  __bf16 v8bf;
typedef __attribute__((ext_vector_type(8)))  float  v8f;

union V16 { v16bf v; v8bf h[2]; };

__device__ inline v8f wmma_bf16(v16bf a, v16bf b, v8f c) {
  return __builtin_amdgcn_wmma_f32_16x16x32_bf16(false, a, false, b,
                                                 (short)0, c, false, false);
}

__device__ inline v8f vzero8() {
  v8f z = {0.f, 0.f, 0.f, 0.f, 0.f, 0.f, 0.f, 0.f};
  return z;
}

__device__ inline float silu_f(float x) {
  return x / (1.f + __expf(-x));
}

// ---------------------------------------------------------------------------
// Weight transpose + fp32 -> bf16 conversion.
// src [rows][cols] fp32 (row-major)  ->  dst [cols][rows] bf16
// ---------------------------------------------------------------------------
__global__ __launch_bounds__(256) void k_cvt_wuvqk(const float* __restrict__ src,
                                                   __bf16* __restrict__ dst) {
  int idx = blockIdx.x * 256 + threadIdx.x;       // E_ * N4_ elements
  int r = idx >> 11;                              // row in [0,512)
  int c = idx & (N4_ - 1);                        // col in [0,2048)
  dst[(size_t)c * E_ + r] = (__bf16)src[idx];
}

__global__ __launch_bounds__(256) void k_cvt_wproj(const float* __restrict__ src,
                                                   __bf16* __restrict__ dst) {
  int idx = blockIdx.x * 256 + threadIdx.x;       // 512*512 elements
  int r = idx >> 9;
  int c = idx & (E_ - 1);
  dst[(size_t)c * E_ + r] = (__bf16)src[idx];
}

// ---------------------------------------------------------------------------
// Input LayerNorm: one wave per row of 512, output bf16.
// ---------------------------------------------------------------------------
__global__ __launch_bounds__(256) void k_ln_in(const float* __restrict__ x,
                                               const float* __restrict__ w,
                                               const float* __restrict__ b,
                                               __bf16* __restrict__ xn) {
  int wv = threadIdx.x >> 5, lane = threadIdx.x & 31;
  int row = blockIdx.x * 8 + wv;
  const float* xr = x + (size_t)row * E_ + lane * 16;
  float vals[16];
  float s = 0.f, ss = 0.f;
#pragma unroll
  for (int i = 0; i < 16; ++i) {
    vals[i] = xr[i];
    s += vals[i];
    ss += vals[i] * vals[i];
  }
#pragma unroll
  for (int off = 16; off >= 1; off >>= 1) {
    s  += __shfl_xor(s,  off, 32);
    ss += __shfl_xor(ss, off, 32);
  }
  float mu   = s * (1.f / (float)E_);
  float var  = ss * (1.f / (float)E_) - mu * mu;
  float rstd = rsqrtf(var + 1e-5f);
  const float* wp = w + lane * 16;
  const float* bp = b + lane * 16;
  __bf16* o = xn + (size_t)row * E_ + lane * 16;
#pragma unroll
  for (int i = 0; i < 16; ++i)
    o[i] = (__bf16)((vals[i] - mu) * rstd * wp[i] + bp[i]);
}

// ---------------------------------------------------------------------------
// GEMM1: mixed = silu(xn @ w_uvqk + b), split into u (f32), q/k ([B,H,S,D]
// bf16), v transposed to [B,H,D,S] bf16.  One wave computes a 16x64 tile.
// waves = (T_/16) * (N4_/64) = 1024 * 32 = 32768 ; 8 waves/block -> 4096 blks
// ---------------------------------------------------------------------------
__global__ __launch_bounds__(256) void k_gemm_uvqk(
    const __bf16* __restrict__ xn, const __bf16* __restrict__ wT,
    const float* __restrict__ bias, float* __restrict__ u_out,
    __bf16* __restrict__ q_out, __bf16* __restrict__ k_out,
    __bf16* __restrict__ vT_out) {
  const int wave = blockIdx.x * 8 + (threadIdx.x >> 5);
  const int lane = threadIdx.x & 31;
  const int mt = wave >> 5;            // 1024 M tiles of 16 rows
  const int ng = wave & 31;            // 32 N groups of 64 cols
  const int row0 = mt * 16;
  const int col0 = ng * 64;
  const int lm = lane & 15;
  const int hs = lane >> 4;

  v8f acc[4];
#pragma unroll
  for (int i = 0; i < 4; ++i) acc[i] = vzero8();

  const __bf16* arow = xn + (size_t)(row0 + lm) * E_;
  for (int k0 = 0; k0 < E_; k0 += 32) {
    V16 a;
    const v8bf* pa = (const v8bf*)(arow + k0 + hs * 8);
    a.h[0] = pa[0];
    a.h[1] = pa[2];  // +16 elements
#pragma unroll
    for (int nt = 0; nt < 4; ++nt) {
      const __bf16* brow = wT + (size_t)(col0 + nt * 16 + lm) * E_ + k0 + hs * 8;
      V16 bb;
      const v8bf* pb = (const v8bf*)brow;
      bb.h[0] = pb[0];
      bb.h[1] = pb[2];
      acc[nt] = wmma_bf16(a.v, bb.v, acc[nt]);
    }
  }

#pragma unroll
  for (int nt = 0; nt < 4; ++nt) {
    int col = col0 + nt * 16 + lm;
    float bn = bias[col];
    int seg = col >> 9;          // 0:u 1:v 2:q 3:k (uniform per wave)
    int cs = col & (E_ - 1);
    int h = cs >> 6, d = cs & 63;
#pragma unroll
    for (int jj = 0; jj < 8; ++jj) {
      int r = row0 + jj + hs * 8;
      float v = silu_f(acc[nt][jj] + bn);
      int bidx = r >> 11, s = r & (S_ - 1);
      if (seg == 0) {
        u_out[(size_t)r * E_ + cs] = v;
      } else if (seg == 1) {
        vT_out[((size_t)(bidx * H_ + h) * D_ + d) * S_ + s] = (__bf16)v;
      } else if (seg == 2) {
        q_out[((size_t)(bidx * H_ + h) * S_ + s) * D_ + d] = (__bf16)v;
      } else {
        k_out[((size_t)(bidx * H_ + h) * S_ + s) * D_ + d] = (__bf16)v;
      }
    }
  }
}

// ---------------------------------------------------------------------------
// SiLU attention with causal + target mask.
// grid = (S_/64, B_*H_), block = 128 (4 waves); each wave owns 16 queries.
// scores = silu(alpha * q k^T)/S, masked; out = scores @ v.
// Scores are staged through per-wave LDS to re-pack C-layout -> A-layout.
// ---------------------------------------------------------------------------
__global__ __launch_bounds__(128) void k_attn(const __bf16* __restrict__ q,
                                              const __bf16* __restrict__ k,
                                              const __bf16* __restrict__ vT,
                                              float* __restrict__ attn_out) {
  __shared__ float sbuf[4][16 * 32];
  const int wv = threadIdx.x >> 5, lane = threadIdx.x & 31;
  const int bh = blockIdx.y;                     // b*H + h
  const int s0 = blockIdx.x * 64 + wv * 16;
  const int lm = lane & 15, hs = lane >> 4;
  const float alpha = 0.125f;                    // 1/sqrt(64)
  const float inv_s = 1.f / (float)S_;

  const __bf16* qp = q + (size_t)bh * S_ * D_;
  const __bf16* kp = k + (size_t)bh * S_ * D_;
  const __bf16* vp = vT + (size_t)bh * D_ * S_;

  // A-matrix: this wave's 16 query rows, two K-slices of 32 (D=64)
  V16 aq[2];
  const __bf16* qrow = qp + (size_t)(s0 + lm) * D_;
#pragma unroll
  for (int sl = 0; sl < 2; ++sl) {
    const v8bf* p = (const v8bf*)(qrow + sl * 32 + hs * 8);
    aq[sl].h[0] = p[0];
    aq[sl].h[1] = p[2];
  }

  v8f acc[4];
#pragma unroll
  for (int i = 0; i < 4; ++i) acc[i] = vzero8();

  const int kbmax = (s0 + 15) >> 5;              // causal: keys <= s0+15
  for (int kb = 0; kb <= kbmax; ++kb) {
    const int j0 = kb * 32;
    // ---- scores: 16 queries x 32 keys, in two 16x16 halves ----
#pragma unroll
    for (int half = 0; half < 2; ++half) {
      const int jc = j0 + half * 16 + lm;        // this lane's key column
      const __bf16* krow = kp + (size_t)jc * D_;
      V16 bk0, bk1;
      {
        const v8bf* p0 = (const v8bf*)(krow + hs * 8);
        bk0.h[0] = p0[0];
        bk0.h[1] = p0[2];
        const v8bf* p1 = (const v8bf*)(krow + 32 + hs * 8);
        bk1.h[0] = p1[0];
        bk1.h[1] = p1[2];
      }
      v8f c = vzero8();
      c = wmma_bf16(aq[0].v, bk0.v, c);
      c = wmma_bf16(aq[1].v, bk1.v, c);
      // silu + scale + mask; write to LDS in [m][key] order
#pragma unroll
      for (int jj = 0; jj < 8; ++jj) {
        int m = jj + hs * 8;
        int iq = s0 + m;
        bool ok = (jc <= iq) && ((jc < HIST_) || (jc == iq));
        float sv = 0.f;
        if (ok) sv = silu_f(c[jj] * alpha) * inv_s;
        sbuf[wv][m * 32 + half * 16 + lm] = sv;
      }
    }
    // same-wave DS ops are in-order; repack scores as bf16 A-matrix
    V16 aat;
#pragma unroll
    for (int t = 0; t < 8; ++t) {
      aat.v[t]     = (__bf16)sbuf[wv][lm * 32 + hs * 8 + t];
      aat.v[t + 8] = (__bf16)sbuf[wv][lm * 32 + 16 + hs * 8 + t];
    }
    // ---- attn @ v over this 32-key block, 4 output d-tiles ----
#pragma unroll
    for (int nd = 0; nd < 4; ++nd) {
      const __bf16* vrow = vp + (size_t)(nd * 16 + lm) * S_ + j0 + hs * 8;
      V16 bv;
      const v8bf* p = (const v8bf*)vrow;
      bv.h[0] = p[0];
      bv.h[1] = p[2];
      acc[nd] = wmma_bf16(aat.v, bv.v, acc[nd]);
    }
  }

  const int bidx = bh >> 3, h = bh & 7;
#pragma unroll
  for (int nd = 0; nd < 4; ++nd) {
#pragma unroll
    for (int jj = 0; jj < 8; ++jj) {
      int srow = s0 + jj + hs * 8;
      size_t t = (size_t)bidx * S_ + srow;
      attn_out[t * (H_ * D_) + h * D_ + nd * 16 + lm] = acc[nd][jj];
    }
  }
}

// ---------------------------------------------------------------------------
// parallel = u * layer_norm(attn_out); one wave per row, output bf16.
// ---------------------------------------------------------------------------
__global__ __launch_bounds__(256) void k_ln_mul(const float* __restrict__ ao,
                                                const float* __restrict__ u,
                                                const float* __restrict__ w,
                                                const float* __restrict__ b,
                                                __bf16* __restrict__ par) {
  int wv = threadIdx.x >> 5, lane = threadIdx.x & 31;
  int row = blockIdx.x * 8 + wv;
  const float* xr = ao + (size_t)row * E_ + lane * 16;
  const float* ur = u + (size_t)row * E_ + lane * 16;
  float vals[16];
  float s = 0.f, ss = 0.f;
#pragma unroll
  for (int i = 0; i < 16; ++i) {
    vals[i] = xr[i];
    s += vals[i];
    ss += vals[i] * vals[i];
  }
#pragma unroll
  for (int off = 16; off >= 1; off >>= 1) {
    s  += __shfl_xor(s,  off, 32);
    ss += __shfl_xor(ss, off, 32);
  }
  float mu   = s * (1.f / (float)E_);
  float var  = ss * (1.f / (float)E_) - mu * mu;
  float rstd = rsqrtf(var + 1e-5f);
  const float* wp = w + lane * 16;
  const float* bp = b + lane * 16;
  __bf16* o = par + (size_t)row * E_ + lane * 16;
#pragma unroll
  for (int i = 0; i < 16; ++i)
    o[i] = (__bf16)(ur[i] * ((vals[i] - mu) * rstd * wp[i] + bp[i]));
}

// ---------------------------------------------------------------------------
// GEMM2: out = parallel @ w_proj + x.  One wave = 16x64 tile.
// waves = 1024 * 8 = 8192 ; 8 waves/block -> 1024 blocks.
// ---------------------------------------------------------------------------
__global__ __launch_bounds__(256) void k_gemm_proj(
    const __bf16* __restrict__ par, const __bf16* __restrict__ wT,
    const float* __restrict__ x, float* __restrict__ out) {
  const int wave = blockIdx.x * 8 + (threadIdx.x >> 5);
  const int lane = threadIdx.x & 31;
  const int mt = wave >> 3;
  const int ng = wave & 7;
  const int row0 = mt * 16;
  const int col0 = ng * 64;
  const int lm = lane & 15;
  const int hs = lane >> 4;

  v8f acc[4];
#pragma unroll
  for (int i = 0; i < 4; ++i) acc[i] = vzero8();

  const __bf16* arow = par + (size_t)(row0 + lm) * E_;
  for (int k0 = 0; k0 < E_; k0 += 32) {
    V16 a;
    const v8bf* pa = (const v8bf*)(arow + k0 + hs * 8);
    a.h[0] = pa[0];
    a.h[1] = pa[2];
#pragma unroll
    for (int nt = 0; nt < 4; ++nt) {
      const __bf16* brow = wT + (size_t)(col0 + nt * 16 + lm) * E_ + k0 + hs * 8;
      V16 bb;
      const v8bf* pb = (const v8bf*)brow;
      bb.h[0] = pb[0];
      bb.h[1] = pb[2];
      acc[nt] = wmma_bf16(a.v, bb.v, acc[nt]);
    }
  }

#pragma unroll
  for (int nt = 0; nt < 4; ++nt) {
    int col = col0 + nt * 16 + lm;
#pragma unroll
    for (int jj = 0; jj < 8; ++jj) {
      int r = row0 + jj + hs * 8;
      out[(size_t)r * E_ + col] = acc[nt][jj] + x[(size_t)r * E_ + col];
    }
  }
}

// ---------------------------------------------------------------------------
// Host launcher
// ---------------------------------------------------------------------------
extern "C" void kernel_launch(void* const* d_in, const int* in_sizes, int n_in,
                              void* d_out, int out_size, void* d_ws,
                              size_t ws_size, hipStream_t stream) {
  const float* x        = (const float*)d_in[0];
  const float* w_uvqk   = (const float*)d_in[1];
  const float* b_uvqk   = (const float*)d_in[2];
  const float* ln_in_w  = (const float*)d_in[3];
  const float* ln_in_b  = (const float*)d_in[4];
  const float* ln_out_w = (const float*)d_in[5];
  const float* ln_out_b = (const float*)d_in[6];
  const float* w_proj   = (const float*)d_in[7];
  float* out = (float*)d_out;

  // workspace carve-out (256B aligned slices)
  size_t off = 0;
  auto alloc = [&](size_t bytes) -> void* {
    void* p = (char*)d_ws + off;
    off += (bytes + 255) & ~(size_t)255;
    return p;
  };
  __bf16* wuvqkT = (__bf16*)alloc((size_t)N4_ * E_ * 2);   //  2 MB
  __bf16* wprojT = (__bf16*)alloc((size_t)E_ * E_ * 2);    //  0.5 MB
  __bf16* xn     = (__bf16*)alloc((size_t)T_ * E_ * 2);    // 16 MB
  float*  u_buf  = (float*) alloc((size_t)T_ * E_ * 4);    // 32 MB
  __bf16* q_buf  = (__bf16*)alloc((size_t)T_ * E_ * 2);    // 16 MB
  __bf16* k_buf  = (__bf16*)alloc((size_t)T_ * E_ * 2);    // 16 MB
  __bf16* vT_buf = (__bf16*)alloc((size_t)T_ * E_ * 2);    // 16 MB
  float*  ao_buf = (float*) alloc((size_t)T_ * E_ * 4);    // 32 MB
  __bf16* par    = (__bf16*)alloc((size_t)T_ * E_ * 2);    // 16 MB
  (void)ws_size; (void)in_sizes; (void)n_in; (void)out_size;

  // 1. weight conversion (transposed, bf16)
  k_cvt_wuvqk<<<(E_ * N4_) / 256, 256, 0, stream>>>(w_uvqk, wuvqkT);
  k_cvt_wproj<<<(E_ * E_) / 256, 256, 0, stream>>>(w_proj, wprojT);

  // 2. input layernorm -> bf16
  k_ln_in<<<T_ / 8, 256, 0, stream>>>(x, ln_in_w, ln_in_b, xn);

  // 3. fused uvqk GEMM + bias + silu + split/transpose
  k_gemm_uvqk<<<(T_ / 16) * (N4_ / 64) / 8, 256, 0, stream>>>(
      xn, wuvqkT, b_uvqk, u_buf, q_buf, k_buf, vT_buf);

  // 4. silu-attention
  dim3 agrid(S_ / 64, B_ * H_);
  k_attn<<<agrid, 128, 0, stream>>>(q_buf, k_buf, vT_buf, ao_buf);

  // 5. u * layernorm(attn_out) -> bf16
  k_ln_mul<<<T_ / 8, 256, 0, stream>>>(ao_buf, u_buf, ln_out_w, ln_out_b, par);

  // 6. projection GEMM + residual
  k_gemm_proj<<<(T_ / 16) * (E_ / 64) / 8, 256, 0, stream>>>(par, wprojT, x, out);
}
